// GraphSAGE_29618094473879
// MI455X (gfx1250) — matrix operand order repaired and checked by reading, not compiled, not run on hardware
//
#include <hip/hip_runtime.h>
#include <math.h>

#define N_NODES 100000
#define N_EDGES 1600000
#define IN_CH   128
#define HID_CH  128
#define OUT_CH  40
#define NPAD    48          // layer-2 N padded to 3 WMMA tiles
#define KTOT    256         // concat K: [agg*invdeg | self]

typedef __attribute__((ext_vector_type(16))) __bf16 v16bf;
typedef __attribute__((ext_vector_type(8)))  __bf16 v8bf;
typedef __attribute__((ext_vector_type(8)))  float  v8f;

// ---------------------------------------------------------------------------
// Degree / inverse-degree
// ---------------------------------------------------------------------------
__global__ void degree_kernel(const long long* __restrict__ dst,
                              float* __restrict__ deg, int nEdges) {
    int e = blockIdx.x * blockDim.x + threadIdx.x;
    if (e < nEdges) atomicAdd(&deg[dst[e]], 1.0f);
}

__global__ void invdeg_kernel(float* __restrict__ deg, int n) {
    int i = blockIdx.x * blockDim.x + threadIdx.x;
    if (i < n) {
        float d = deg[i];
        deg[i] = d > 0.0f ? 1.0f / d : 0.0f;
    }
}

// ---------------------------------------------------------------------------
// Scatter-sum of 128-ch features: 32 lanes/edge, b128 gather + 4 f32 atomics
// ---------------------------------------------------------------------------
__global__ __launch_bounds__(256)
void scatter_kernel(const float* __restrict__ feat,
                    const long long* __restrict__ src,
                    const long long* __restrict__ dst,
                    float* __restrict__ agg, int nEdges) {
    int t = blockIdx.x * blockDim.x + threadIdx.x;
    int e = t >> 5;
    if (e >= nEdges) return;
    int lane = t & 31;
    long long s = src[e];
    long long d = dst[e];
    const float4 v = *reinterpret_cast<const float4*>(feat + (size_t)s * 128 + lane * 4);
    float* dp = agg + (size_t)d * 128 + lane * 4;
    atomicAdd(dp + 0, v.x);
    atomicAdd(dp + 1, v.y);
    atomicAdd(dp + 2, v.z);
    atomicAdd(dp + 3, v.w);
}

// ---------------------------------------------------------------------------
// Weight prep: transpose + convert to bf16, N-major (Wt[n][k], k=0..255)
// Wt1: 128 x 256 ; Wt2: 48 x 256 (cols 40..47 zero) -> B loads become two
// contiguous global_load_b128 per fragment.
// ---------------------------------------------------------------------------
__global__ void prep_weights(const float* __restrict__ W1l, const float* __restrict__ W1r,
                             const float* __restrict__ W2l, const float* __restrict__ W2r,
                             __bf16* __restrict__ Wt1, __bf16* __restrict__ Wt2) {
    int i = blockIdx.x * blockDim.x + threadIdx.x;
    if (i < HID_CH * KTOT) {
        int n = i >> 8, k = i & 255;
        float v = (k < 128) ? W1l[(size_t)k * HID_CH + n]
                            : W1r[(size_t)(k - 128) * HID_CH + n];
        Wt1[i] = (__bf16)v;
    }
    if (i < NPAD * KTOT) {
        int n = i >> 8, k = i & 255;
        float v = 0.0f;
        if (n < OUT_CH)
            v = (k < 128) ? W2l[(size_t)k * OUT_CH + n]
                          : W2r[(size_t)(k - 128) * OUT_CH + n];
        Wt2[i] = (__bf16)v;
    }
}

// Deterministic dropout keep mask (stand-in for jax PRNG stream)
__device__ inline bool keep_fn(unsigned idx) {
    unsigned x = idx * 0x9E3779B9u + 0x42424242u;
    x ^= x >> 16; x *= 0x7feb352du;
    x ^= x >> 15; x *= 0x846ca68bu;
    x ^= x >> 16;
    return (x & 1u) != 0u;
}

#define AS_STRIDE 264   // 528B row stride: rows land on disjoint 4-bank groups

// Build one A fragment (16x32 bf16, ISA 7.12.2 layout) from the LDS stage
__device__ inline v16bf a_frag_from_lds(const __bf16* As_row, int kt, int khalf) {
    int kb = kt * 32 + khalf * 8;
    v8bf a0 = *(const v8bf*)(As_row + kb);
    v8bf a1 = *(const v8bf*)(As_row + kb + 16);
    v16bf a;
#pragma unroll
    for (int j = 0; j < 8; ++j) { a[j] = a0[j]; a[8 + j] = a1[j]; }
    return a;
}

// Build one B fragment (32x16 bf16) from the N-major bf16 weight buffer
__device__ inline v16bf b_frag_from_wt(const __bf16* wcol, int kt, int khalf) {
    int kb = kt * 32 + khalf * 8;
    v8bf b0 = *(const v8bf*)(wcol + kb);
    v8bf b1 = *(const v8bf*)(wcol + kb + 16);
    v16bf b;
#pragma unroll
    for (int j = 0; j < 8; ++j) { b[j] = b0[j]; b[8 + j] = b1[j]; }
    return b;
}

// ---------------------------------------------------------------------------
// Layer 1: h = dropout(relu([agg*invdeg | x] @ Wt1^T + b1))
// Block: 256 threads = 8 waves, one 16-row m-tile, waves own the 8 n-tiles.
// A staged once in LDS (bf16, invdeg pre-applied); K=256 in 8 WMMA steps.
// ---------------------------------------------------------------------------
__global__ __launch_bounds__(256)
void gemm1_kernel(const float* __restrict__ agg, const float* __restrict__ invdeg,
                  const float* __restrict__ x,
                  const __bf16* __restrict__ Wt1,
                  const float* __restrict__ b1, float* __restrict__ h) {
    __shared__ __bf16 As[16][AS_STRIDE];
    int tid = threadIdx.x;
    int m0  = blockIdx.x * 16;

    {   // cooperative stage: unit = (row, 16-wide K segment); 256 units, 1/thread
        int row  = tid >> 4;
        int seg  = tid & 15;
        int node = m0 + row;
        const float* srcp;
        float s;
        if (seg < 8) { srcp = agg + (size_t)node * 128 + seg * 16;      s = invdeg[node]; }
        else         { srcp = x   + (size_t)node * 128 + (seg - 8) * 16; s = 1.0f; }
#pragma unroll
        for (int j = 0; j < 16; ++j)
            As[row][seg * 16 + j] = (__bf16)(srcp[j] * s);
    }
    __syncthreads();

    int lane  = tid & 31;
    int n0    = (tid >> 5) * 16;
    int rc    = lane & 15;
    int khalf = lane >> 4;
    const __bf16* As_row = &As[rc][0];
    const __bf16* wcol   = Wt1 + (size_t)(n0 + rc) * KTOT;

    v8f acc = {};
#pragma unroll
    for (int kt = 0; kt < 8; ++kt) {
        v16bf a = a_frag_from_lds(As_row, kt, khalf);
        v16bf b = b_frag_from_wt(wcol, kt, khalf);
        acc = __builtin_amdgcn_wmma_f32_16x16x32_bf16(
            false, a, false, b, (short)0, acc, false, false);
    }

    int n = n0 + rc;
    float bias = b1[n];
#pragma unroll
    for (int r = 0; r < 8; ++r) {
        int row = m0 + khalf * 8 + r;
        float v = acc[r] + bias;
        v = v > 0.0f ? v : 0.0f;
        v = keep_fn((unsigned)row * 128u + (unsigned)n) ? v * 2.0f : 0.0f;
        h[(size_t)row * 128 + n] = v;
    }
}

// ---------------------------------------------------------------------------
// Layer 2: out = [agg2*invdeg | h] @ Wt2^T + b2   (logits -> d_out)
// Block: 96 threads = 3 waves = the 3 padded n-tiles of one 16-row m-tile.
// ---------------------------------------------------------------------------
__global__ __launch_bounds__(96)
void gemm2_kernel(const float* __restrict__ agg, const float* __restrict__ invdeg,
                  const float* __restrict__ h,
                  const __bf16* __restrict__ Wt2,
                  const float* __restrict__ b2, float* __restrict__ out) {
    __shared__ __bf16 As[16][AS_STRIDE];
    int tid = threadIdx.x;
    int m0  = blockIdx.x * 16;

    for (int u = tid; u < 256; u += 96) {   // 256 stage units over 96 threads
        int row  = u >> 4;
        int seg  = u & 15;
        int node = m0 + row;
        const float* srcp;
        float s;
        if (seg < 8) { srcp = agg + (size_t)node * 128 + seg * 16;      s = invdeg[node]; }
        else         { srcp = h   + (size_t)node * 128 + (seg - 8) * 16; s = 1.0f; }
#pragma unroll
        for (int j = 0; j < 16; ++j)
            As[row][seg * 16 + j] = (__bf16)(srcp[j] * s);
    }
    __syncthreads();

    int lane  = tid & 31;
    int n0    = (tid >> 5) * 16;
    int rc    = lane & 15;
    int khalf = lane >> 4;
    const __bf16* As_row = &As[rc][0];
    const __bf16* wcol   = Wt2 + (size_t)(n0 + rc) * KTOT;

    v8f acc = {};
#pragma unroll
    for (int kt = 0; kt < 8; ++kt) {
        v16bf a = a_frag_from_lds(As_row, kt, khalf);
        v16bf b = b_frag_from_wt(wcol, kt, khalf);
        acc = __builtin_amdgcn_wmma_f32_16x16x32_bf16(
            false, a, false, b, (short)0, acc, false, false);
    }

    int n = n0 + rc;
    bool cv = n < OUT_CH;
    float bias = cv ? b2[cv ? n : 0] : 0.0f;
#pragma unroll
    for (int r = 0; r < 8; ++r) {
        int row = m0 + khalf * 8 + r;
        if (cv) out[(size_t)row * OUT_CH + n] = acc[r] + bias;
    }
}

// ---------------------------------------------------------------------------
// In-place log_softmax over 40 classes, one thread per row
// ---------------------------------------------------------------------------
__global__ void logsoftmax_kernel(float* __restrict__ out, int n) {
    int i = blockIdx.x * blockDim.x + threadIdx.x;
    if (i >= n) return;
    float* p = out + (size_t)i * OUT_CH;
    float v[OUT_CH];
    float m = -INFINITY;
#pragma unroll
    for (int j = 0; j < OUT_CH; ++j) {
        v[j] = p[j];
        m = v[j] > m ? v[j] : m;
    }
    float s = 0.0f;
#pragma unroll
    for (int j = 0; j < OUT_CH; ++j) s += __expf(v[j] - m);
    float ls = __logf(s);
#pragma unroll
    for (int j = 0; j < OUT_CH; ++j) p[j] = v[j] - m - ls;
}

// ---------------------------------------------------------------------------
extern "C" void kernel_launch(void* const* d_in, const int* in_sizes, int n_in,
                              void* d_out, int out_size, void* d_ws, size_t ws_size,
                              hipStream_t stream) {
    (void)in_sizes; (void)n_in; (void)out_size; (void)ws_size;

    const float*     x   = (const float*)d_in[0];
    const long long* ei  = (const long long*)d_in[1];   // int64 edge_index
    const float*     W1l = (const float*)d_in[2];
    const float*     b1  = (const float*)d_in[3];
    const float*     W1r = (const float*)d_in[4];
    const float*     W2l = (const float*)d_in[5];
    const float*     b2  = (const float*)d_in[6];
    const float*     W2r = (const float*)d_in[7];
    float*           out = (float*)d_out;

    const long long* src = ei;
    const long long* dst = ei + N_EDGES;

    // workspace layout: deg/invdeg | agg (reused both layers) | h | Wt1 | Wt2
    float*  deg = (float*)d_ws;
    float*  agg = deg + N_NODES;
    float*  h   = agg + (size_t)N_NODES * HID_CH;
    __bf16* Wt1 = (__bf16*)(h + (size_t)N_NODES * HID_CH);
    __bf16* Wt2 = Wt1 + (size_t)HID_CH * KTOT;

    const size_t aggBytes = (size_t)N_NODES * HID_CH * sizeof(float);

    hipMemsetAsync(deg, 0, N_NODES * sizeof(float), stream);
    hipMemsetAsync(agg, 0, aggBytes, stream);

    prep_weights<<<(HID_CH * KTOT + 255) / 256, 256, 0, stream>>>(
        W1l, W1r, W2l, W2r, Wt1, Wt2);

    degree_kernel<<<(N_EDGES + 255) / 256, 256, 0, stream>>>(dst, deg, N_EDGES);
    invdeg_kernel<<<(N_NODES + 255) / 256, 256, 0, stream>>>(deg, N_NODES);

    // layer 1 aggregation + GEMM
    scatter_kernel<<<(N_EDGES + 7) / 8, 256, 0, stream>>>(x, src, dst, agg, N_EDGES);
    gemm1_kernel<<<N_NODES / 16, 256, 0, stream>>>(agg, deg, x, Wt1, b1, h);

    // layer 2 aggregation + GEMM (logits straight into d_out)
    hipMemsetAsync(agg, 0, aggBytes, stream);
    scatter_kernel<<<(N_EDGES + 7) / 8, 256, 0, stream>>>(h, src, dst, agg, N_EDGES);
    gemm2_kernel<<<N_NODES / 16, 96, 0, stream>>>(agg, deg, h, Wt2, b2, out);

    logsoftmax_kernel<<<(N_NODES + 255) / 256, 256, 0, stream>>>(out, N_NODES);
}
